// TreeLSTM_39857296507718
// MI455X (gfx1250) — compile-verified, matrix operand order checked by reading, not run on hardware
//
#include <hip/hip_runtime.h>

// CDNA5 (gfx1250) TreeLSTM, v3.
// - One-time pack: x -> bf16, {W_iou|U_iou|U_f} -> unified-K bf16 weight buffer (1280 x 768).
// - Per level: bf16 WMMA GEMM with DOUBLE-BUFFERED async global->LDS staging
//   (global_load_async_to_lds_b128 / ASYNCcnt), 4 wmma/wave/K-step, 1 barrier/K-step.
// - Gate kernel writes h as fp32 (d_out) and bf16 (workspace, feeds parent level).

typedef __attribute__((ext_vector_type(16))) __bf16 v16bf;
typedef __attribute__((ext_vector_type(8)))  float  v8f;

#define BM 64
#define BN 128
#define BK 32
#define NCOLS_FULL 1280   // 768 iou + 512 f
#define KTOT_FULL  768    // 256 (x) + 512 (h_cat)

__device__ __forceinline__ unsigned short f2bf(float f) {
    union { float f; unsigned u; } v; v.f = f;
    unsigned u = v.u;
    unsigned r = (u + 0x7FFFu + ((u >> 16) & 1u)) >> 16;  // RNE truncation
    return (unsigned short)r;
}

__device__ __forceinline__ float sigf(float x) {
    return 1.0f / (1.0f + __expf(-x));
}

// 16B global -> LDS async copy (ASYNCcnt). lds_dst: pointer into __shared__;
// low 32 bits of the flat address are the LDS byte offset (aperture is the high dword).
__device__ __forceinline__ void async_copy_b128(unsigned short* lds_dst,
                                                const unsigned short* gsrc) {
    unsigned lds_off = (unsigned)(size_t)lds_dst;
    asm volatile("global_load_async_to_lds_b128 %0, %1, off"
                 :: "v"(lds_off), "v"(gsrc)
                 : "memory");
}

__device__ __forceinline__ void wait_async0() {
#if __has_builtin(__builtin_amdgcn_s_wait_asynccnt)
    __builtin_amdgcn_s_wait_asynccnt(0);
#else
    asm volatile("s_wait_asynccnt 0x0" ::: "memory");
#endif
}

// ---------- one-time packers ----------

__global__ __launch_bounds__(256)
void pack_bf16(const float* __restrict__ s, unsigned short* __restrict__ d, int n4)
{
    const int i = blockIdx.x * 256 + threadIdx.x;
    if (i >= n4) return;
    const float4 f = ((const float4*)s)[i];
    unsigned short* o = d + (size_t)i * 4;
    o[0] = f2bf(f.x); o[1] = f2bf(f.y); o[2] = f2bf(f.z); o[3] = f2bf(f.w);
}

// Unified-K weight buffer Bw[n][kg], n in [0,1280), kg in [0,768):
//   n <  768: kg<256 -> W_iou[n][kg], else U_iou[n][kg-256]
//   n >= 768: kg<256 -> 0,            else U_f[n-768][kg-256]
__global__ __launch_bounds__(256)
void build_wb(const float* __restrict__ W_iou, const float* __restrict__ U_iou,
              const float* __restrict__ U_f, unsigned short* __restrict__ Bw)
{
    const int i = blockIdx.x * 256 + threadIdx.x;
    if (i >= NCOLS_FULL * KTOT_FULL) return;
    const int n  = i / KTOT_FULL;
    const int kg = i - n * KTOT_FULL;
    float v = 0.0f;
    if (n < 768) v = (kg < 256) ? W_iou[(size_t)n * 256 + kg]
                                : U_iou[(size_t)n * 512 + (kg - 256)];
    else if (kg >= 256) v = U_f[(size_t)(n - 768) * 512 + (kg - 256)];
    Bw[i] = f2bf(v);
}

// ---------- level GEMM (bf16 operands, fp32 accum, async double-buffered LDS) ----------
// A row p: kg<256 -> xb[(node_off+p)*256 + kg]; kg>=256 -> hb_ch[p*512 + (kg-256)]
// B row n: Bw[n*768 + kg]
__global__ __launch_bounds__(256)
void tlstm_level_gemm(const unsigned short* __restrict__ xb,
                      const unsigned short* __restrict__ hb_ch,
                      const unsigned short* __restrict__ Bw,
                      float* __restrict__ S,
                      int m, int node_off, int n_cols, int k_total)
{
    __shared__ unsigned short Abuf[2][BM * BK];  // [buf][row*BK + k]
    __shared__ unsigned short Bbuf[2][BN * BK];  // [buf][col*BK + k]

    const int tid  = threadIdx.x;
    const int wave = tid >> 5;
    const int lane = tid & 31;
    const int half = lane >> 4;
    const int idx  = lane & 15;

    const int r0 = blockIdx.x * BM;
    const int c0 = blockIdx.y * BN;

    const int mw = wave & 3;   // 16-row strip within 64
    const int nw = wave >> 2;  // 64-col half within 128

    v8f acc[4] = {v8f{}, v8f{}, v8f{}, v8f{}};

    // per-thread staging coordinates (constant over K loop)
    const int alin  = tid << 3;           // 2048 A elems, one b128 per thread
    const int arow  = alin >> 5;
    const int akk   = alin & (BK - 1);
    const int agrow = r0 + arow;
    const bool a_ok = (agrow < m);

    const int b0lin  = tid << 3;          // first of two b128 B copies per thread
    const int b0row  = b0lin >> 5;
    const int b0kk   = b0lin & (BK - 1);
    const int b1lin  = (tid + 256) << 3;
    const int b1row  = b1lin >> 5;
    const int b1kk   = b1lin & (BK - 1);

    const int steps = k_total / BK;

    // stage tile `kb` into buffer `buf`
    auto issue_tile = [&](int kb, int buf) {
        if (a_ok) {
            const int kg = kb + akk;      // region-uniform across the 8 elems
            const unsigned short* src = (kg < 256)
                ? (xb + (size_t)(node_off + agrow) * 256 + kg)
                : (hb_ch + (size_t)agrow * 512 + (kg - 256));
            async_copy_b128(&Abuf[buf][arow * BK + akk], src);
            __builtin_prefetch(src + 2 * BK, 0, 1);   // hint 2 K-tiles ahead
        }
        async_copy_b128(&Bbuf[buf][b0row * BK + b0kk],
                        Bw + (size_t)(c0 + b0row) * KTOT_FULL + kb + b0kk);
        async_copy_b128(&Bbuf[buf][b1row * BK + b1kk],
                        Bw + (size_t)(c0 + b1row) * KTOT_FULL + kb + b1kk);
    };

    issue_tile(0, 0);

    for (int i = 0; i < steps; ++i) {
        wait_async0();        // my async writes of buf[i&1] have landed
        __syncthreads();      // everyone's writes landed; prior reads of other buf done
        if (i + 1 < steps) issue_tile((i + 1) * BK, (i + 1) & 1);

        const unsigned short* Ab = Abuf[i & 1];
        const unsigned short* Bb = Bbuf[i & 1];

        union Frag { uint4 q[2]; v16bf v; } fa;
        {
            const unsigned short* ap = Ab + (mw * 16 + idx) * BK + half * 8;
            fa.q[0] = *(const uint4*)(ap);
            fa.q[1] = *(const uint4*)(ap + 16);
        }
#pragma unroll
        for (int j = 0; j < 4; ++j) {
            union Frag fb;
            const unsigned short* bp = Bb + (nw * 64 + j * 16 + idx) * BK + half * 8;
            fb.q[0] = *(const uint4*)(bp);
            fb.q[1] = *(const uint4*)(bp + 16);
            acc[j] = __builtin_amdgcn_wmma_f32_16x16x32_bf16(false, fa.v, false, fb.v,
                                                             (short)0, acc[j], false, false);
        }
    }

    // ---- store: D VGPR r -> row (r + 8*half), col = idx ----
    const int col0 = c0 + nw * 64;
#pragma unroll
    for (int r = 0; r < 8; ++r) {
        const int grow = r0 + mw * 16 + half * 8 + r;
        if (grow < m) {
            float* dst = S + (size_t)grow * n_cols + col0;
#pragma unroll
            for (int j = 0; j < 4; ++j)
                dst[j * 16 + idx] = acc[j][r];
        }
    }
}

// ---------- gate / cell-update ----------
__global__ __launch_bounds__(256)
void tlstm_gate(const float* __restrict__ S,
                const float* __restrict__ b_iou,
                const float* __restrict__ U_f_b,
                const float* __restrict__ c_ch,   // child-level c block, or null
                float* __restrict__ h_out,
                unsigned short* __restrict__ hb_out,
                float* __restrict__ c_out,
                int m, int node_off, int stride, int has_children)
{
    const int g = blockIdx.x * blockDim.x + threadIdx.x;
    if (g >= m * 256) return;
    const int p = g >> 8;
    const int j = g & 255;

    const float* s = S + (size_t)p * stride;
    const float pi = s[j]       + b_iou[j];
    const float po = s[256 + j] + b_iou[256 + j];
    const float pu = s[512 + j] + b_iou[512 + j];

    float c_in = 0.0f;
    if (has_children) {
        const float pf0 = s[768 + j]  + U_f_b[j];
        const float pf1 = s[1024 + j] + U_f_b[256 + j];
        c_in = sigf(pf0) * c_ch[(size_t)(2 * p) * 256 + j]
             + sigf(pf1) * c_ch[(size_t)(2 * p + 1) * 256 + j];
    }
    const float cn = sigf(pi) * tanhf(pu) + c_in;
    const float hn = sigf(po) * tanhf(cn);

    const size_t o = (size_t)(node_off + p) * 256 + j;
    c_out[o]  = cn;
    h_out[o]  = hn;
    hb_out[o] = f2bf(hn);   // bf16 copy feeds the parent level's WMMA A-operand
}

extern "C" void kernel_launch(void* const* d_in, const int* in_sizes, int n_in,
                              void* d_out, int out_size, void* d_ws, size_t ws_size,
                              hipStream_t stream) {
    (void)in_sizes; (void)n_in; (void)out_size; (void)ws_size;
    const int DEPTH = 17;
    const int N = (1 << DEPTH) - 1;

    const float* x     = (const float*)d_in[0];
    const float* W_iou = (const float*)d_in[3];
    const float* U_iou = (const float*)d_in[4];
    const float* b_iou = (const float*)d_in[5];
    const float* U_f_w = (const float*)d_in[6];
    const float* U_f_b = (const float*)d_in[7];

    float* h = (float*)d_out;   // heap-layout h == final output

    // workspace carve-up
    char* ws = (char*)d_ws;
    float* c = (float*)ws;                                  ws += (size_t)N * 256 * 4;
    float* S = (float*)ws;                                  ws += (size_t)65536 * 768 * 4;
    unsigned short* xb = (unsigned short*)ws;               ws += (size_t)N * 256 * 2;
    unsigned short* hb = (unsigned short*)ws;               ws += (size_t)N * 256 * 2;
    unsigned short* Bw = (unsigned short*)ws;

    // --- one-time packs ---
    {
        const int n4 = (N * 256) / 4;
        pack_bf16<<<(n4 + 255) / 256, 256, 0, stream>>>(x, xb, n4);
        const int nw_el = NCOLS_FULL * KTOT_FULL;
        build_wb<<<(nw_el + 255) / 256, 256, 0, stream>>>(W_iou, U_iou, U_f_w, Bw);
    }

    // --- leaves: level DEPTH-1, iou = x@W_iou^T only (K=256, cols=768) ---
    {
        const int m = 1 << (DEPTH - 1);
        const int off = m - 1;
        dim3 grid((m + BM - 1) / BM, 768 / BN);
        tlstm_level_gemm<<<grid, 256, 0, stream>>>(xb, nullptr, Bw, S, m, off, 768, 256);
        tlstm_gate<<<(m * 256 + 255) / 256, 256, 0, stream>>>(
            S, b_iou, U_f_b, nullptr, h, hb, c, m, off, 768, 0);
    }

    // --- internal levels, bottom-up ---
    for (int k = DEPTH - 2; k >= 0; --k) {
        const int m     = 1 << k;
        const int off   = m - 1;
        const int choff = 2 * m - 1;   // first child row (children are contiguous pairs)
        dim3 grid((m + BM - 1) / BM, NCOLS_FULL / BN);
        tlstm_level_gemm<<<grid, 256, 0, stream>>>(
            xb, hb + (size_t)choff * 256, Bw, S, m, off, NCOLS_FULL, KTOT_FULL);
        tlstm_gate<<<(m * 256 + 255) / 256, 256, 0, stream>>>(
            S, b_iou, U_f_b, c + (size_t)choff * 256, h, hb, c, m, off, NCOLS_FULL, 1);
    }
}
